// GATv2_ten_89704686944359
// MI455X (gfx1250) — matrix operand order repaired and checked by reading, not compiled, not run on hardware
//
#include <hip/hip_runtime.h>
#include <math.h>
#include <stdint.h>

// GATv2, 10 layers, N=50000, E=800000 (+N self loops), H=4.
// Dense lin_l/lin_r GEMMs via V_WMMA_F32_16X16X32_F16 (f32 accumulate),
// with weights prepacked to f16 WMMA-fragment order (1 vector load / fragment).
// Edge phase (coalesced wave-per-edge gather, segment softmax, atomic scatter).

typedef __attribute__((ext_vector_type(16))) _Float16 v16h;
typedef __attribute__((ext_vector_type(8)))  float    v8f;

#define HEADS 4

__device__ __forceinline__ float lrelu(float v) { return v > 0.f ? v : 0.2f * v; }

__device__ __forceinline__ float atomicMaxF(float* addr, float val) {
  int* ai = (int*)addr;
  int old = __float_as_int(*addr);
  while (__int_as_float(old) < val) {
    int assumed = old;
    old = atomicCAS(ai, assumed, __float_as_int(val));
    if (old == assumed) break;
  }
  return __int_as_float(old);
}

__global__ void fill_kernel(float* __restrict__ p, float v, int n) {
  int i = blockIdx.x * blockDim.x + threadIdx.x;
  if (i < n) p[i] = v;
}

// Pack W[K,OC] (f32, row-major) into f16 WMMA B-fragment order:
// fragment index frag = colTile*(K/32) + kc ; per lane one contiguous v16h.
// B per-lane k order used here (contiguous, matches A): col = ct*16 + lane%16, K = kc*32 + (lane/16)*16 + 2v.
__global__ void pack_w_dual(const float* __restrict__ Wl, const float* __restrict__ Wr,
                            _Float16* __restrict__ pkL, _Float16* __restrict__ pkR,
                            int K, int OC) {
  int t = blockIdx.x * blockDim.x + threadIdx.x;
  int kchunks = K >> 5;
  int frags = (OC >> 4) * kchunks;
  if (t >= frags * 32) return;
  int lane = t & 31;
  int frag = t >> 5;
  int kc = frag % kchunks;
  int ct = frag / kchunks;
  int half = lane >> 4;
  int col = ct * 16 + (lane & 15);
  v16h vl, vr;
#pragma unroll
  for (int v = 0; v < 8; ++v) {
    int kb = kc * 32 + (half << 4) + (v << 1);
    vl[2 * v]     = (_Float16)Wl[(size_t)kb * OC + col];
    vl[2 * v + 1] = (_Float16)Wl[(size_t)(kb + 1) * OC + col];
    vr[2 * v]     = (_Float16)Wr[(size_t)kb * OC + col];
    vr[2 * v + 1] = (_Float16)Wr[(size_t)(kb + 1) * OC + col];
  }
  ((v16h*)pkL)[t] = vl;
  ((v16h*)pkR)[t] = vr;
}

// One wave computes one 16x16 tile of BOTH xl and xr (shared A fragment).
__global__ void gemm_dual_wmma(const float* __restrict__ h,
                               const _Float16* __restrict__ pkL, const float* __restrict__ bl,
                               const _Float16* __restrict__ pkR, const float* __restrict__ br,
                               float* __restrict__ xl, float* __restrict__ xr,
                               int n, int K, int OC, int rowTiles, int colTiles) {
  int wave = blockIdx.x * (blockDim.x >> 5) + (threadIdx.x >> 5);
  if (wave >= rowTiles * colTiles) return;
  int rowTile = wave % rowTiles;
  int colTile = wave / rowTiles;
  int lane = threadIdx.x & 31;
  int half = lane >> 4;
  int r16  = lane & 15;
  int rowBase = rowTile * 16;
  int col = colTile * 16 + r16;            // OC is a multiple of 16
  int arow = rowBase + r16;
  if (arow >= n) arow = n - 1;             // clamp: keep EXEC all-ones for WMMA

  int kchunks = K >> 5;
  const v16h* fl = (const v16h*)pkL + (size_t)colTile * kchunks * 32 + lane;
  const v16h* fr = (const v16h*)pkR + (size_t)colTile * kchunks * 32 + lane;

  v8f cl = {0.f, 0.f, 0.f, 0.f, 0.f, 0.f, 0.f, 0.f};
  v8f cr = cl;

  for (int kc = 0; kc < kchunks; ++kc) {
    v16h a;
    const float* hrow = h + (size_t)arow * K + kc * 32;
#pragma unroll
    for (int v = 0; v < 8; ++v) {
      // A k order used here (contiguous, matches packed B): K = half*16 + 2v
      int ka = (half << 4) + (v << 1);
      float2 av = *(const float2*)(hrow + ka);
      a[2 * v]     = (_Float16)av.x;
      a[2 * v + 1] = (_Float16)av.y;
    }
    v16h bL = fl[(size_t)kc * 32];
    v16h bR = fr[(size_t)kc * 32];
    cl = __builtin_amdgcn_wmma_f32_16x16x32_f16(false, a, false, bL, (short)0, cl, false, false);
    cr = __builtin_amdgcn_wmma_f32_16x16x32_f16(false, a, false, bR, (short)0, cr, false, false);
  }

  float blv = bl[col];
  float brv = br[col];
  if (rowBase + 16 <= n) {                 // full tile: no per-element guards
#pragma unroll
    for (int d = 0; d < 8; ++d) {
      int row = rowBase + d + (half << 3); // C/D: VGPR d -> M = d + 8*half
      xl[(size_t)row * OC + col] = cl[d] + blv;
      xr[(size_t)row * OC + col] = cr[d] + brv;
    }
  } else {
#pragma unroll
    for (int d = 0; d < 8; ++d) {
      int row = rowBase + d + (half << 3);
      if (row < n) {
        xl[(size_t)row * OC + col] = cl[d] + blv;
        xr[(size_t)row * OC + col] = cr[d] + brv;
      }
    }
  }
}

// Wave-per-edge: lanes cover H*OC channels -> coalesced row gathers, then
// per-head wave reduction via shfl_xor. logit + segment max over dst.
__global__ void edge_logits(const float* __restrict__ xl, const float* __restrict__ xr,
                            const float* __restrict__ att,
                            const int* __restrict__ esrc, const int* __restrict__ edst,
                            float* __restrict__ logit, float* __restrict__ mbuf,
                            int E, int Etot, int OC) {
  int wave = blockIdx.x * (blockDim.x >> 5) + (threadIdx.x >> 5);
  if (wave >= Etot) return;
  int lane = threadIdx.x & 31;
  int s = (wave < E) ? esrc[wave] : (wave - E);
  int d = (wave < E) ? edst[wave] : (wave - E);
  int tot = HEADS * OC;
  const float* xls = xl + (size_t)s * tot;
  const float* xrd = xr + (size_t)d * tot;
  float p0 = 0.f, p1 = 0.f, p2 = 0.f, p3 = 0.f;
  for (int ch = lane; ch < tot; ch += 32) {
    int hh = ch / OC;
    float v = lrelu(xls[ch] + xrd[ch]) * att[ch];   // att flat == [h*OC + c]
    p0 += (hh == 0) ? v : 0.f;
    p1 += (hh == 1) ? v : 0.f;
    p2 += (hh == 2) ? v : 0.f;
    p3 += (hh == 3) ? v : 0.f;
  }
#pragma unroll
  for (int off = 16; off > 0; off >>= 1) {
    p0 += __shfl_xor(p0, off);
    p1 += __shfl_xor(p1, off);
    p2 += __shfl_xor(p2, off);
    p3 += __shfl_xor(p3, off);
  }
  if (lane == 0) {
    float pv[HEADS] = {p0, p1, p2, p3};
#pragma unroll
    for (int hh = 0; hh < HEADS; ++hh) {
      logit[(size_t)wave * HEADS + hh] = pv[hh];
      atomicMaxF(&mbuf[(size_t)d * HEADS + hh], pv[hh]);
    }
  }
}

// p = exp(logit - m[dst]) stored back into logit; segment sum over dst.
__global__ void edge_expsum(float* __restrict__ logit, const float* __restrict__ mbuf,
                            float* __restrict__ sbuf, const int* __restrict__ edst,
                            int E, int Etot) {
  int e = blockIdx.x * blockDim.x + threadIdx.x;
  if (e >= Etot) return;
  int d = (e < E) ? edst[e] : (e - E);
#pragma unroll
  for (int hh = 0; hh < HEADS; ++hh) {
    float p = __expf(logit[(size_t)e * HEADS + hh] - mbuf[(size_t)d * HEADS + hh]);
    logit[(size_t)e * HEADS + hh] = p;
    atomicAdd(&sbuf[(size_t)d * HEADS + hh], p);
  }
}

// acc[dst, h, c] += (p/s[dst,h]) * xl[src, h, c].  Wave-per-edge, lane-coalesced.
__global__ void edge_scatter(const float* __restrict__ xl, const float* __restrict__ p,
                             const float* __restrict__ sbuf,
                             const int* __restrict__ esrc, const int* __restrict__ edst,
                             float* __restrict__ accb, int E, int Etot, int OC) {
  int wave = blockIdx.x * (blockDim.x >> 5) + (threadIdx.x >> 5);
  if (wave >= Etot) return;
  int lane = threadIdx.x & 31;
  int s = (wave < E) ? esrc[wave] : (wave - E);
  int d = (wave < E) ? edst[wave] : (wave - E);
  int tot = HEADS * OC;
  for (int ch = lane; ch < tot; ch += 32) {
    int hh = ch / OC;
    float alpha = p[(size_t)wave * HEADS + hh] / sbuf[(size_t)d * HEADS + hh];
    atomicAdd(&accb[(size_t)d * tot + ch], alpha * xl[(size_t)s * tot + ch]);
  }
}

// Layers 1..9: h = lrelu(acc + bias)   (HC == 32)
__global__ void post_act(const float* __restrict__ accb, const float* __restrict__ bias,
                         float* __restrict__ hout, int n32) {
  int i = blockIdx.x * blockDim.x + threadIdx.x;
  if (i < n32) hout[i] = lrelu(accb[i] + bias[i & 31]);
}

// Layer 10: mean over heads + bias + log_softmax -> d_out[N,40]
__global__ void final_logsoftmax(const float* __restrict__ accb, const float* __restrict__ bias,
                                 float* __restrict__ out, int n) {
  int i = blockIdx.x * blockDim.x + threadIdx.x;
  if (i >= n) return;
  float t[40];
  float mx = -1e30f;
#pragma unroll
  for (int c = 0; c < 40; ++c) {
    float v = 0.f;
#pragma unroll
    for (int hh = 0; hh < HEADS; ++hh) v += accb[(size_t)i * 160 + hh * 40 + c];
    v = v * 0.25f + bias[c];
    t[c] = v;
    if (v > mx) mx = v;
  }
  float se = 0.f;
#pragma unroll
  for (int c = 0; c < 40; ++c) se += __expf(t[c] - mx);
  float lse = mx + __logf(se);
#pragma unroll
  for (int c = 0; c < 40; ++c) out[(size_t)i * 40 + c] = t[c] - lse;
}

static void run_gemm(const float* h, const float* Wl, const float* bl,
                     const float* Wr, const float* br,
                     _Float16* pkL, _Float16* pkR,
                     float* xl, float* xr, int N, int K, int OC, hipStream_t stream) {
  int frags = (OC / 16) * (K / 32);
  int pthreads = frags * 32;
  pack_w_dual<<<(pthreads + 255) / 256, 256, 0, stream>>>(Wl, Wr, pkL, pkR, K, OC);
  int rowTiles = (N + 15) / 16;
  int colTiles = OC / 16;
  int waves = rowTiles * colTiles;
  int blocks = (waves + 7) / 8;            // 8 waves (256 threads) per block
  gemm_dual_wmma<<<blocks, 256, 0, stream>>>(h, pkL, bl, pkR, br, xl, xr,
                                             N, K, OC, rowTiles, colTiles);
}

static void run_edges(const float* xl, const float* xr, const float* att,
                      const int* esrc, const int* edst,
                      float* logit, float* mbuf, float* sbuf, float* accb,
                      int N, int E, int Etot, int OC, hipStream_t stream) {
  int n4 = N * HEADS;
  fill_kernel<<<(n4 + 255) / 256, 256, 0, stream>>>(mbuf, -1e30f, n4);
  fill_kernel<<<(n4 + 255) / 256, 256, 0, stream>>>(sbuf, 0.f, n4);
  int accn = N * HEADS * OC;
  fill_kernel<<<(accn + 255) / 256, 256, 0, stream>>>(accb, 0.f, accn);
  int gw = (Etot + 7) / 8;                 // wave-per-edge
  edge_logits<<<gw, 256, 0, stream>>>(xl, xr, att, esrc, edst, logit, mbuf, E, Etot, OC);
  int ge = (Etot + 255) / 256;
  edge_expsum<<<ge, 256, 0, stream>>>(logit, mbuf, sbuf, edst, E, Etot);
  edge_scatter<<<gw, 256, 0, stream>>>(xl, logit, sbuf, esrc, edst, accb, E, Etot, OC);
}

extern "C" void kernel_launch(void* const* d_in, const int* in_sizes, int n_in,
                              void* d_out, int out_size, void* d_ws, size_t ws_size,
                              hipStream_t stream) {
  const int F = 128, HC = 32, NC = 40;
  const int N = in_sizes[0] / F;
  const int E = in_sizes[1] / 2;
  const int Etot = E + N;

  const float* x   = (const float*)d_in[0];
  const int*   ei  = (const int*)d_in[1];
  const int* esrc = ei;
  const int* edst = ei + E;
  const float* W1l = (const float*)d_in[2];
  const float* b1l = (const float*)d_in[3];
  const float* W1r = (const float*)d_in[4];
  const float* b1r = (const float*)d_in[5];
  const float* att1  = (const float*)d_in[6];
  const float* bias1 = (const float*)d_in[7];
  const float* Wml = (const float*)d_in[8];    // [8,32,32]
  const float* bml = (const float*)d_in[9];    // [8,32]
  const float* Wmr = (const float*)d_in[10];
  const float* bmr = (const float*)d_in[11];
  const float* attm  = (const float*)d_in[12]; // [8,4,8]
  const float* biasm = (const float*)d_in[13]; // [8,32]
  const float* W10l = (const float*)d_in[14];  // [32,160]
  const float* b10l = (const float*)d_in[15];
  const float* W10r = (const float*)d_in[16];
  const float* b10r = (const float*)d_in[17];
  const float* att10  = (const float*)d_in[18]; // [4,40]
  const float* bias10 = (const float*)d_in[19]; // [40]

  // Workspace layout (f32), ~124 MB total + small f16 weight-pack area
  float* ws    = (float*)d_ws;
  float* h0    = ws;                              // N*32
  float* h1    = h0 + (size_t)N * 32;             // N*32
  float* xl    = h1 + (size_t)N * 32;             // N*160 (max width)
  float* xr    = xl + (size_t)N * 160;            // N*160
  float* accb  = xr + (size_t)N * 160;            // N*160
  float* logit = accb + (size_t)N * 160;          // Etot*4
  float* mbuf  = logit + (size_t)Etot * 4;        // N*4
  float* sbuf  = mbuf + (size_t)N * 4;            // N*4
  uintptr_t pw = (uintptr_t)(sbuf + (size_t)N * 4);
  pw = (pw + 31) & ~(uintptr_t)31;                // 32B align for v16h loads
  _Float16* pkL = (_Float16*)pw;                  // 8192 halfs
  _Float16* pkR = pkL + 8192;                     // 8192 halfs

  // ---- Layer 1: F=128 -> 32 ----
  run_gemm(x, W1l, b1l, W1r, b1r, pkL, pkR, xl, xr, N, F, HC, stream);
  run_edges(xl, xr, att1, esrc, edst, logit, mbuf, sbuf, accb, N, E, Etot, 8, stream);
  post_act<<<(N * 32 + 255) / 256, 256, 0, stream>>>(accb, bias1, h0, N * 32);

  // ---- Layers 2..9: 32 -> 32 ----
  float* hc = h0;
  float* hn = h1;
  for (int i = 0; i < 8; ++i) {
    run_gemm(hc, Wml + (size_t)i * HC * HC, bml + (size_t)i * HC,
             Wmr + (size_t)i * HC * HC, bmr + (size_t)i * HC,
             pkL, pkR, xl, xr, N, HC, HC, stream);
    run_edges(xl, xr, attm + (size_t)i * HEADS * 8, esrc, edst,
              logit, mbuf, sbuf, accb, N, E, Etot, 8, stream);
    post_act<<<(N * 32 + 255) / 256, 256, 0, stream>>>(accb, biasm + (size_t)i * HC, hn, N * 32);
    float* t = hc; hc = hn; hn = t;
  }

  // ---- Layer 10: 32 -> H*NC=160, mean over heads, log_softmax ----
  run_gemm(hc, W10l, b10l, W10r, b10r, pkL, pkR, xl, xr, N, HC, HEADS * NC, stream);
  run_edges(xl, xr, att10, esrc, edst, logit, mbuf, sbuf, accb, N, E, Etot, NC, stream);
  final_logsoftmax<<<(N + 255) / 256, 256, 0, stream>>>(accb, bias10, (float*)d_out, N);
}
